// GGNN_47132971107215
// MI455X (gfx1250) — matrix operand order repaired and checked by reading, not compiled, not run on hardware
//
#include <hip/hip_runtime.h>

#define N_V   2048
#define DEG   64
#define NE    (N_V * DEG)      // 131072
#define HID   256
#define ITERS 10

typedef __attribute__((ext_vector_type(16))) _Float16 v16h;
typedef __attribute__((ext_vector_type(8)))  _Float16 v8h;
typedef __attribute__((ext_vector_type(8)))  float    v8f;

// ---------------------------------------------------------------------------
// agent-scope (DEV) 8-byte load/store for cross-step coherence inside the
// single-wave sweep (bypasses WGP$ per CDNA5 SCOPE rules)
// ---------------------------------------------------------------------------
__device__ __forceinline__ float2 agent_load2(const float2* p) {
    unsigned long long v = __hip_atomic_load((const unsigned long long*)p,
                                             __ATOMIC_RELAXED, __HIP_MEMORY_SCOPE_AGENT);
    float2 r;
    __builtin_memcpy(&r, &v, 8);
    return r;
}
__device__ __forceinline__ void agent_store2(float2* p, float2 d) {
    unsigned long long v;
    __builtin_memcpy(&v, &d, 8);
    __hip_atomic_store((unsigned long long*)p, v,
                       __ATOMIC_RELAXED, __HIP_MEMORY_SCOPE_AGENT);
}

__device__ __forceinline__ float lse2(float x, float y) {
    float m = fmaxf(x, y);
    return m + __logf(__expf(x - m) + __expf(y - m));
}

// ---------------------------------------------------------------------------
// Sequential Gauss-Seidel sweep: ONE wave32 walks all 2048 vertices.
// Messages window (vertices i-32..i+32) lives in a 128-vertex LDS ring;
// a depth-PF register queue prefetches vertex i+33 from global well ahead
// of its ds_store so the chain never waits on L2.
// ---------------------------------------------------------------------------
#define PF 8
__global__ __launch_bounds__(32) void sweep_kernel(float2* __restrict__ msg,
                                                   const int* __restrict__ rev,
                                                   const float* __restrict__ Jw,
                                                   const float* __restrict__ b) {
    __shared__ float2 ring[8192];   // 128 vertices * 64 edges = 64KB
    const int lane = threadIdx.x;

    // initial window: vertices 2016..2047 (old) and 0..32 (old)
    for (int t = 0; t < 65; ++t) {
        int v = (2016 + t) & (N_V - 1);
        int e = v * DEG;
        ring[(e & 8191) + lane]      = agent_load2(&msg[e + lane]);
        ring[(e & 8191) + lane + 32] = agent_load2(&msg[e + lane + 32]);
    }

    // prefetch queue warm-up: vertices 33..33+PF-1
    float2 qa[PF], qb[PF];
#pragma unroll
    for (int k = 0; k < PF; ++k) {
        int e = (33 + k) * DEG;
        qa[k] = agent_load2(&msg[e + lane]);
        qb[k] = agent_load2(&msg[e + lane + 32]);
    }

    for (int o = 0; o < N_V / PF; ++o) {
#pragma unroll
        for (int k = 0; k < PF; ++k) {
            const int i    = o * PF + k;
            const int base = i * DEG;

            int e0 = rev[base + lane];
            int e1 = rev[base + lane + 32];
            float2 inc0 = ring[e0 & 8191];
            float2 inc1 = ring[e1 & 8191];

            float sx = inc0.x + inc1.x;
            float sy = inc0.y + inc1.y;
#pragma unroll
            for (int m = 16; m >= 1; m >>= 1) {
                sx += __shfl_xor(sx, m);
                sy += __shfl_xor(sy, m);
            }

            const float bi = b[i];
            const float J0 = Jw[base + lane];
            const float J1 = Jw[base + lane + 32];

            // local[i,j,a,c] = Jw*xij[a,c] + b_i*xi[a], xi={-1,+1}, xij=[[1,-1],[-1,1]]
            float p0x = sx - inc0.x, p0y = sy - inc0.y;
            float2 n0;
            n0.x = lse2(p0x + J0 - bi, p0y - J0 + bi);
            n0.y = lse2(p0x - J0 - bi, p0y + J0 + bi);
            float p1x = sx - inc1.x, p1y = sy - inc1.y;
            float2 n1;
            n1.x = lse2(p1x + J1 - bi, p1y - J1 + bi);
            n1.y = lse2(p1x - J1 - bi, p1y + J1 + bi);

            ring[(base & 8191) + lane]      = n0;
            ring[(base & 8191) + lane + 32] = n1;
            agent_store2(&msg[base + lane],      n0);
            agent_store2(&msg[base + lane + 32], n1);

            // commit prefetched vertex i+33 into the ring
            int ec = ((i + 33) & (N_V - 1)) * DEG;
            ring[(ec & 8191) + lane]      = qa[k];
            ring[(ec & 8191) + lane + 32] = qb[k];

            // issue next prefetch (vertex i+33+PF), PF steps of slack
            int en = ((i + 33 + PF) & (N_V - 1)) * DEG;
            qa[k] = agent_load2(&msg[en + lane]);
            qb[k] = agent_load2(&msg[en + lane + 32]);
        }
    }
}

// ---------------------------------------------------------------------------
// probs[v][c] = b[v]*xi[c] + sum_j msg[rev[v*64+j]][c]   (one wave per vertex)
// ---------------------------------------------------------------------------
__global__ __launch_bounds__(256) void probs_kernel(const float2* __restrict__ msg,
                                                    const int* __restrict__ rev,
                                                    const float* __restrict__ b,
                                                    float2* __restrict__ probs) {
    const int lane = threadIdx.x & 31;
    const int v    = blockIdx.x * 8 + (threadIdx.x >> 5);
    const int base = v * DEG;
    float2 a = msg[rev[base + lane]];
    float2 c = msg[rev[base + lane + 32]];
    float sx = a.x + c.x, sy = a.y + c.y;
#pragma unroll
    for (int m = 16; m >= 1; m >>= 1) {
        sx += __shfl_xor(sx, m);
        sy += __shfl_xor(sy, m);
    }
    if (lane == 0) {
        float bv = b[v];
        probs[v] = make_float2(sx - bv, sy + bv);
    }
}

// ---------------------------------------------------------------------------
// Pack W2 (256x256 f32 row-major) into B-fragment layout for
// v_wmma_f32_16x16x32_f16: lane L holds column n=L&15, K-range
// kt*32 + (t>=8?16:0) + (L>=16?8:0) + (t&7)  (16 halves per lane -> 2x b128 loads)
// ---------------------------------------------------------------------------
__global__ __launch_bounds__(256) void pack_w2_kernel(const float* __restrict__ W2,
                                                      _Float16* __restrict__ W2p) {
    int idx  = blockIdx.x * 256 + threadIdx.x;     // 65536 total
    int t    = idx & 15;
    int lane = (idx >> 4) & 31;
    int nt   = (idx >> 9) & 15;
    int kt   = idx >> 13;
    int k = kt * 32 + (t >> 3) * 16 + (lane >> 4) * 8 + (t & 7);
    int n = nt * 16 + (lane & 15);
    W2p[idx] = (_Float16)W2[k * HID + n];
}

// ---------------------------------------------------------------------------
// Calibration MLP, 64 edge-rows per block (4 waves of wave32).
//   layer1 (K=4) in VALU -> h1 f16 in LDS (A-fragment friendly, row-major)
//   layer2 (256x256) via v_wmma_f32_16x16x32_f16, A frags hoisted per kt
//   layer3 (dot with W3) fused into WMMA epilogue + 16-lane shuffle reduce
//   sigmoid gate -> damped message written to outNew
// ---------------------------------------------------------------------------
__global__ __launch_bounds__(128) void calib_kernel(const float2* __restrict__ msg,
                                                    const float2* __restrict__ oldm,
                                                    const float2* __restrict__ probs,
                                                    const int* __restrict__ rev,
                                                    const float* __restrict__ W1,
                                                    const float* __restrict__ b1,
                                                    const _Float16* __restrict__ W2p,
                                                    const float* __restrict__ b2,
                                                    const float* __restrict__ W3,
                                                    const float* __restrict__ b3,
                                                    const float* __restrict__ m0new,
                                                    float* __restrict__ outNew,
                                                    int state) {
    __shared__ __align__(16) float    Xs[64][4];
    __shared__ __align__(16) _Float16 h1[64 * HID];   // 32KB

    const int tid      = threadIdx.x;
    const int blockRow = blockIdx.x * 64;

    // ---- features: X = [m_s, o_s, probs[src,s], f3] -------------------------
    if (tid < 64) {
        int e = blockRow + tid;
        float2 m = msg[e], o = oldm[e];
        float2 p = probs[e >> 6];
        int re = rev[e];
        float f0, f1, f2, f3;
        if (state == 0) {
            float2 mr = msg[re];
            f0 = m.x; f1 = o.x; f2 = p.x; f3 = mr.x + mr.y;
        } else {
            f0 = m.y; f1 = o.y; f2 = p.y; f3 = msg[re].y + m0new[re];
        }
        Xs[tid][0] = f0; Xs[tid][1] = f1; Xs[tid][2] = f2; Xs[tid][3] = f3;
    }
    __syncthreads();

    // ---- layer 1: h1 = relu(X @ W1 + b1), K=4 (VALU) ------------------------
#pragma unroll
    for (int hh = 0; hh < 2; ++hh) {
        int n = tid + hh * 128;
        float w0 = W1[n], w1 = W1[HID + n], w2 = W1[2 * HID + n], w3 = W1[3 * HID + n];
        float bb = b1[n];
        for (int m = 0; m < 64; ++m) {
            float4 x = *reinterpret_cast<const float4*>(&Xs[m][0]);
            float h = fmaf(x.x, w0, fmaf(x.y, w1, fmaf(x.z, w2, fmaf(x.w, w3, bb))));
            h1[m * HID + n] = (_Float16)fmaxf(h, 0.f);
        }
    }
    __syncthreads();

    // ---- layers 2+3: WMMA 16x16x32 f16, fused W3 dot ------------------------
    const int lane = tid & 31;
    const int l16  = lane & 15;
    const int hi   = lane >> 4;
    const int rowBase = (tid >> 5) * 16;   // wave owns 16 rows

    // A fragments for this wave's 16 rows (depend only on kt) — per ISA layout:
    // lanes 0-15: K = kt*32 + {0..7, 16..23}; lanes 16-31: +8
    v16h afr[8];
#pragma unroll
    for (int kt = 0; kt < 8; ++kt) {
        const _Float16* ar = &h1[(rowBase + l16) * HID + kt * 32 + hi * 8];
        v8h alo = *reinterpret_cast<const v8h*>(ar);
        v8h ahi = *reinterpret_cast<const v8h*>(ar + 16);
        afr[kt] = __builtin_shufflevector(alo, ahi, 0, 1, 2, 3, 4, 5, 6, 7,
                                                    8, 9, 10, 11, 12, 13, 14, 15);
    }

    float p3[8];
#pragma unroll
    for (int r = 0; r < 8; ++r) p3[r] = 0.f;

    for (int nt = 0; nt < 16; ++nt) {
        v8f acc = {0.f, 0.f, 0.f, 0.f, 0.f, 0.f, 0.f, 0.f};
#pragma unroll
        for (int kt = 0; kt < 8; ++kt) {
            const v8h* bp = reinterpret_cast<const v8h*>(&W2p[((kt * 16 + nt) * 32 + lane) * 16]);
            v8h b0 = bp[0], b1v = bp[1];
            v16h bfr = __builtin_shufflevector(b0, b1v, 0, 1, 2, 3, 4, 5, 6, 7,
                                                        8, 9, 10, 11, 12, 13, 14, 15);
            acc = __builtin_amdgcn_wmma_f32_16x16x32_f16(false, afr[kt], false, bfr,
                                                         (short)0, acc, false, false);
        }
        int n = nt * 16 + l16;
        float b2n = b2[n], w3n = W3[n];
#pragma unroll
        for (int r = 0; r < 8; ++r)             // acc[r] = h2[rowBase + r + 8*hi][n]
            p3[r] += fmaxf(acc[r] + b2n, 0.f) * w3n;
    }

    // reduce over the 16 lanes sharing the same row set
#pragma unroll
    for (int r = 0; r < 8; ++r) {
        float v = p3[r];
        v += __shfl_xor(v, 1);
        v += __shfl_xor(v, 2);
        v += __shfl_xor(v, 4);
        v += __shfl_xor(v, 8);
        p3[r] = v;
    }

    if (l16 == 0) {
        float bb3 = b3[0];
#pragma unroll
        for (int r = 0; r < 8; ++r) {
            int e = blockRow + rowBase + hi * 8 + r;
            float z = p3[r] + bb3;
            float a = 1.f / (1.f + __expf(-z));
            float2 m = msg[e], o = oldm[e];
            float mv = state ? m.y : m.x;
            float ov = state ? o.y : o.x;
            outNew[e] = (1.f - a) * mv + a * ov;
        }
    }
}

// ---------------------------------------------------------------------------
__global__ __launch_bounds__(256) void commit_kernel(float2* __restrict__ msg,
                                                     const float* __restrict__ m0n,
                                                     const float* __restrict__ m1n) {
    int e = blockIdx.x * 256 + threadIdx.x;
    msg[e] = make_float2(m0n[e], m1n[e]);
}

__global__ __launch_bounds__(256) void softmax_kernel(const float2* __restrict__ probs,
                                                      float* __restrict__ out) {
    int v = blockIdx.x * 256 + threadIdx.x;
    if (v < N_V) {
        float2 p = probs[v];
        float m = fmaxf(p.x, p.y);
        float e0 = __expf(p.x - m), e1 = __expf(p.y - m);
        float s = e0 + e1;
        out[2 * v]     = e0 / s;
        out[2 * v + 1] = e1 / s;
    }
}

// ---------------------------------------------------------------------------
extern "C" void kernel_launch(void* const* d_in, const int* in_sizes, int n_in,
                              void* d_out, int out_size, void* d_ws, size_t ws_size,
                              hipStream_t stream) {
    const float* b  = (const float*)d_in[0];
    const float* Jw = (const float*)d_in[1];
    const int*   rv = (const int*)d_in[2];
    const float* W1 = (const float*)d_in[3];
    const float* b1 = (const float*)d_in[4];
    const float* W2 = (const float*)d_in[5];
    const float* b2 = (const float*)d_in[6];
    const float* W3 = (const float*)d_in[7];
    const float* b3 = (const float*)d_in[8];

    char* ws = (char*)d_ws;
    float2*   msg   = (float2*)(ws);                               // 1 MB
    float2*   oldm  = (float2*)(ws + (1u << 20));                  // 1 MB
    float*    m0n   = (float*)(ws + (2u << 20));                   // 512 KB
    float*    m1n   = (float*)(ws + (2u << 20) + (512u << 10));    // 512 KB
    float2*   probs = (float2*)(ws + (3u << 20));                  // 16 KB
    _Float16* W2p   = (_Float16*)(ws + (3u << 20) + (64u << 10));  // 128 KB

    hipMemsetAsync(msg, 0, NE * sizeof(float2), stream);           // messages start at 0
    pack_w2_kernel<<<256, 256, 0, stream>>>(W2, W2p);

    for (int it = 0; it < ITERS; ++it) {
        hipMemcpyAsync(oldm, msg, NE * sizeof(float2),
                       hipMemcpyDeviceToDevice, stream);
        sweep_kernel<<<1, 32, 0, stream>>>(msg, rv, Jw, b);
        probs_kernel<<<N_V / 8, 256, 0, stream>>>(msg, rv, b, probs);
        calib_kernel<<<NE / 64, 128, 0, stream>>>(msg, oldm, probs, rv,
                                                  W1, b1, W2p, b2, W3, b3,
                                                  m0n, m0n, 0);
        calib_kernel<<<NE / 64, 128, 0, stream>>>(msg, oldm, probs, rv,
                                                  W1, b1, W2p, b2, W3, b3,
                                                  m0n, m1n, 1);
        commit_kernel<<<NE / 256, 256, 0, stream>>>(msg, m0n, m1n);
    }
    probs_kernel<<<N_V / 8, 256, 0, stream>>>(msg, rv, b, probs);
    softmax_kernel<<<8, 256, 0, stream>>>(probs, (float*)d_out);
}
